// CWS_35682588295359
// MI455X (gfx1250) — compile-verified, matrix operand order checked by reading, not run on hardware
//
#include <hip/hip_runtime.h>
#include <hip/hip_bf16.h>

#define BB    64
#define SS    512
#define EMB   256
#define HID   256
#define H2    128
#define GG    512        // 4*H2 gates
#define NTAGS 4

typedef __attribute__((ext_vector_type(16))) __bf16 v16bf;
typedef __attribute__((ext_vector_type(8)))  __bf16 v8bf;
typedef __attribute__((ext_vector_type(8)))  float  v8f;

// ---------------- helpers ----------------
__device__ __forceinline__ __bf16 to_bf16(float f) {
  unsigned u = __builtin_bit_cast(unsigned, f);
  unsigned lsb = (u >> 16) & 1u;
  u += 0x7FFFu + lsb;                      // round-to-nearest-even
  unsigned short h = (unsigned short)(u >> 16);
  return __builtin_bit_cast(__bf16, h);
}

__device__ __forceinline__ float sigmoidf_(float x) {
  return 1.0f / (1.0f + __expf(-x));
}

// Load one 16x32 bf16 WMMA A/B fragment for this lane from a row-major row.
// Per CDNA5 ISA 7.12.2 (16-bit A 16x32): lane<16 holds K = {kh*8+0..7, 16+kh*8+0..7}
// where kh = lane>>4.  Two contiguous 8-element (16B) chunks -> b128 loads.
__device__ __forceinline__ v16bf load_frag(const __bf16* row, int kbase, int kh) {
  v8bf lo = *(const v8bf*)(row + kbase + kh * 8);
  v8bf hi = *(const v8bf*)(row + kbase + 16 + kh * 8);
  v16bf r;
#pragma unroll
  for (int i = 0; i < 8; ++i) { r[i] = lo[i]; r[i + 8] = hi[i]; }
  return r;
}

// ---------------- prep kernels ----------------
__global__ void k_embed(const int* __restrict__ sent, const float* __restrict__ emb,
                        __bf16* __restrict__ xbf) {
  int tid = blockIdx.x * blockDim.x + threadIdx.x;      // B*S*EMB threads
  int bs = tid >> 8, e = tid & 255;
  int tok = sent[bs];
  xbf[tid] = to_bf16(emb[(size_t)tok * EMB + e]);
}

__global__ void k_cvt_w(const float* __restrict__ wih, const float* __restrict__ whh,
                        __bf16* __restrict__ wihb, __bf16* __restrict__ whhb) {
  const int NIH = 2 * 2 * GG * EMB;                     // 524288
  const int NHH = 2 * 2 * GG * H2;                      // 262144
  int tid = blockIdx.x * blockDim.x + threadIdx.x;
  if (tid < NIH)            wihb[tid] = to_bf16(wih[tid]);
  else if (tid < NIH + NHH) whhb[tid - NIH] = to_bf16(whh[tid - NIH]);
}

__global__ void k_f2b(const float* __restrict__ src, __bf16* __restrict__ dst, int n) {
  int tid = blockIdx.x * blockDim.x + threadIdx.x;
  if (tid < n) dst[tid] = to_bf16(src[tid]);
}

// ---------------- batched gate GEMM:  XW = X(bf16, MxK) * W^T(bf16, GGxK) + bias ----------------
// M = B*S = 32768, N = GG = 512, K = EMB = 256.
// One wave computes a 16x64 strip (4 N-tiles): A fragment reused 4x, 32 WMMAs/wave.
__global__ void __launch_bounds__(256) k_gemm(const __bf16* __restrict__ X,
                                              const __bf16* __restrict__ W,
                                              const float* __restrict__ bih,
                                              const float* __restrict__ bhh,
                                              float* __restrict__ out) {
  int lane = threadIdx.x & 31, wv = threadIdx.x >> 5;
  int tile = blockIdx.x * 8 + wv;    // 16384 wave-strips
  int mt = tile >> 3;                // 0..2047 (M tiles)
  int ng = tile & 7;                 // 0..7    (N groups of 4 tiles = 64 cols)
  int ln = lane & 15, kh = lane >> 4;

  const __bf16* arow  = X + (size_t)(mt * 16 + ln) * EMB;       // A: row m, k contiguous
  const __bf16* brow0 = W + (size_t)(ng * 64 + ln) * EMB;       // B: W rows (= W^T cols)
  v8f acc[4] = {v8f{}, v8f{}, v8f{}, v8f{}};
#pragma unroll
  for (int k = 0; k < EMB; k += 32) {
    v16bf a = load_frag(arow, k, kh);
#pragma unroll
    for (int q = 0; q < 4; ++q) {
      v16bf b = load_frag(brow0 + (size_t)q * 16 * EMB, k, kh);
      acc[q] = __builtin_amdgcn_wmma_f32_16x16x32_bf16(false, a, false, b, (short)0,
                                                       acc[q], false, false);
    }
  }
#pragma unroll
  for (int q = 0; q < 4; ++q) {
    int n = ng * 64 + q * 16 + ln;
    float bias = bih[n] + bhh[n];
#pragma unroll
    for (int r = 0; r < 8; ++r) {
      int m = mt * 16 + r + kh * 8;                       // C/D layout: vgpr r, halves +8
      __builtin_nontemporal_store(acc[q][r] + bias, &out[(size_t)m * GG + n]);
    }
  }
}

// ---------------- persistent recurrent LSTM scan ----------------
// grid.x = 2 (direction). 8 waves; wave jc owns hidden cols [jc*16, jc*16+16).
// Whh fragments resident in VGPRs; h kept as bf16 in LDS (padded rows); h/c f32 in registers.
#define HROW 136   // padded LDS row stride (bf16 elems), 16B-aligned, bank-spread

__global__ void __launch_bounds__(256) k_lstm(const float* __restrict__ gates0,
                                              const float* __restrict__ gates1,
                                              const __bf16* __restrict__ whhb,  // [2][GG][H2]
                                              const int* __restrict__ length,
                                              float* __restrict__ feats) {
  const int dir = blockIdx.x;
  const float* __restrict__ gates = dir ? gates1 : gates0;
  const __bf16* __restrict__ W = whhb + (size_t)dir * GG * H2;

  __shared__ __align__(16) __bf16 hbuf[64 * HROW];
  __shared__ int Ls[BB];

  int lane = threadIdx.x & 31, jc = threadIdx.x >> 5;
  int ln = lane & 15, kh = lane >> 4;

  for (int i = threadIdx.x; i < 64 * HROW; i += 256) hbuf[i] = to_bf16(0.0f);
  if (threadIdx.x < BB) Ls[threadIdx.x] = length[threadIdx.x];
  __syncthreads();

  // Resident B fragments of Whh: gate group g, k-step kk.
  v16bf Bf[4][4];
#pragma unroll
  for (int g = 0; g < 4; ++g) {
    const __bf16* brow = W + (size_t)(g * H2 + jc * 16 + ln) * H2;
#pragma unroll
    for (int kk = 0; kk < 4; ++kk) Bf[g][kk] = load_frag(brow, kk * 32, kh);
  }

  float hreg[4][8], creg[4][8];
#pragma unroll
  for (int mt = 0; mt < 4; ++mt)
#pragma unroll
    for (int r = 0; r < 8; ++r) { hreg[mt][r] = 0.0f; creg[mt][r] = 0.0f; }

  const int j = jc * 16 + ln;

  for (int t = 0; t < SS; ++t) {
    // Prefetch next step's 128KB of gate rows (global_prefetch_b8) so the L2->WGP
    // stream overlaps this step's WMMA + transcendental work.
    {
      int tn = t + 1;
      if (tn < SS) {
#pragma unroll
        for (int p = 0; p < 4; ++p) {
          int idx = (int)threadIdx.x + p * 256;   // 0..1023
          int b = idx >> 4;                       // 64 batch rows
          int seg = idx & 15;                     // 16 x 128B segments per row
          int L = Ls[b];
          int pos = dir ? ((tn < L) ? (L - 1 - tn) : tn) : tn;
          __builtin_prefetch(&gates[((size_t)b * SS + pos) * GG + seg * 32], 0, 1);
        }
      }
    }
#pragma unroll
    for (int mt = 0; mt < 4; ++mt) {
      // accumulator = precomputed x@Wih.T + bias  (single-use: nontemporal loads)
      v8f acc[4];
#pragma unroll
      for (int g = 0; g < 4; ++g) {
#pragma unroll
        for (int r = 0; r < 8; ++r) {
          int b = mt * 16 + r + kh * 8;
          int L = Ls[b];
          int pos = dir ? ((t < L) ? (L - 1 - t) : t) : t;
          acc[g][r] =
              __builtin_nontemporal_load(&gates[((size_t)b * SS + pos) * GG + g * H2 + j]);
        }
      }
      // h @ Whh.T via WMMA
#pragma unroll
      for (int kk = 0; kk < 4; ++kk) {
        v16bf a = load_frag(hbuf + (mt * 16 + ln) * HROW, kk * 32, kh);
#pragma unroll
        for (int g = 0; g < 4; ++g)
          acc[g] = __builtin_amdgcn_wmma_f32_16x16x32_bf16(false, a, false, Bf[g][kk],
                                                           (short)0, acc[g], false, false);
      }
      // gate nonlinearities + mask blend, all in the native D layout (pure per-lane VALU)
#pragma unroll
      for (int r = 0; r < 8; ++r) {
        int b = mt * 16 + r + kh * 8;
        int L = Ls[b];
        bool valid = (t < L);
        float iv = sigmoidf_(acc[0][r]);
        float fv = sigmoidf_(acc[1][r]);
        float gv = tanhf(acc[2][r]);
        float ov = sigmoidf_(acc[3][r]);
        float c2 = fv * creg[mt][r] + iv * gv;
        float h2 = ov * tanhf(c2);
        float hn = valid ? h2 : hreg[mt][r];
        float cn = valid ? c2 : creg[mt][r];
        hreg[mt][r] = hn;
        creg[mt][r] = cn;
        int pos = dir ? ((t < L) ? (L - 1 - t) : t) : t;
        __builtin_nontemporal_store(valid ? h2 : 0.0f,
                                    &feats[((size_t)b * SS + pos) * HID + dir * H2 + j]);
      }
    }
    __syncthreads();   // all reads of hbuf for step t done
#pragma unroll
    for (int mt = 0; mt < 4; ++mt)
#pragma unroll
      for (int r = 0; r < 8; ++r)
        hbuf[(mt * 16 + r + kh * 8) * HROW + j] = to_bf16(hreg[mt][r]);
    __syncthreads();   // new h visible
  }
}

// ---------------- tag projection: em = feats @ w_tag.T + b_tag ----------------
__global__ void k_em(const float* __restrict__ feats, const float* __restrict__ wtag,
                     const float* __restrict__ btag, float* __restrict__ em) {
  int tid = blockIdx.x * blockDim.x + threadIdx.x;    // B*S*NTAGS
  if (tid >= BB * SS * NTAGS) return;
  int tag = tid & 3, bs = tid >> 2;
  const float4* f = (const float4*)(feats + (size_t)bs * HID);
  const float4* w = (const float4*)(wtag + (size_t)tag * HID);
  float s = btag[tag];
#pragma unroll 4
  for (int k = 0; k < HID / 4; ++k) {
    float4 fv = f[k], wv = w[k];
    s += fv.x * wv.x + fv.y * wv.y + fv.z * wv.z + fv.w * wv.w;
  }
  em[tid] = s;
}

// ---------------- CRF: gold score + log-partition + mean NLL ----------------
__global__ void k_crf(const float* __restrict__ em, const int* __restrict__ tags,
                      const int* __restrict__ length, const float* __restrict__ startt,
                      const float* __restrict__ endt, const float* __restrict__ trans,
                      float* __restrict__ out) {
  __shared__ float red[BB];
  int b = threadIdx.x;
  const float* e0 = em + (size_t)b * SS * NTAGS;
  const int* tg = tags + (size_t)b * SS;
  int L = length[b];

  float tr[16], st[4], en[4];
#pragma unroll
  for (int i = 0; i < 16; ++i) tr[i] = trans[i];
#pragma unroll
  for (int i = 0; i < 4; ++i) { st[i] = startt[i]; en[i] = endt[i]; }

  float alpha[4];
#pragma unroll
  for (int jj = 0; jj < 4; ++jj) alpha[jj] = st[jj] + e0[jj];

  float score = st[tg[0]] + e0[tg[0]];
  for (int t = 1; t < SS; ++t) {
    if (t < L) {
      const float* e = e0 + t * NTAGS;
      float nxt[4];
#pragma unroll
      for (int jj = 0; jj < 4; ++jj) {
        float m = alpha[0] + tr[0 * 4 + jj];
#pragma unroll
        for (int i = 1; i < 4; ++i) m = fmaxf(m, alpha[i] + tr[i * 4 + jj]);
        float s = 0.0f;
#pragma unroll
        for (int i = 0; i < 4; ++i) s += __expf(alpha[i] + tr[i * 4 + jj] - m);
        nxt[jj] = m + __logf(s) + e[jj];
      }
#pragma unroll
      for (int jj = 0; jj < 4; ++jj) alpha[jj] = nxt[jj];
      score += tr[tg[t - 1] * 4 + tg[t]] + e[tg[t]];
    }
  }
  score += en[tg[L - 1]];

  float m = alpha[0] + en[0];
#pragma unroll
  for (int i = 1; i < 4; ++i) m = fmaxf(m, alpha[i] + en[i]);
  float s = 0.0f;
#pragma unroll
  for (int i = 0; i < 4; ++i) s += __expf(alpha[i] + en[i] - m);
  float logz = m + __logf(s);

  red[b] = score - logz;
  __syncthreads();
  if (b == 0) {
    float acc = 0.0f;
    for (int i = 0; i < BB; ++i) acc += red[i];
    out[0] = -acc / (float)BB;
  }
}

// ---------------- workspace layout (bytes, 256B-aligned) ----------------
#define OFF_XBF   ((size_t)0)                         // B*S*EMB bf16      = 16 MB
#define OFF_G0    ((size_t)16777216)                  // B*S*GG f32 dir0   = 64 MB
#define OFF_G1    ((size_t)83886080)                  // B*S*GG f32 dir1   = 64 MB
#define OFF_FEATS ((size_t)150994944)                 // B*S*HID f32       = 32 MB
#define OFF_EM    ((size_t)184549376)                 // B*S*NTAGS f32     = 0.5 MB
#define OFF_WIH   ((size_t)185073664)                 // 2*2*GG*EMB bf16   = 1 MB
#define OFF_WHH   ((size_t)186122240)                 // 2*2*GG*H2 bf16    = 0.5 MB

extern "C" void kernel_launch(void* const* d_in, const int* in_sizes, int n_in,
                              void* d_out, int out_size, void* d_ws, size_t ws_size,
                              hipStream_t stream) {
  const int*   sentence    = (const int*)d_in[0];
  const int*   tags        = (const int*)d_in[1];
  const int*   length      = (const int*)d_in[3];
  const float* embed       = (const float*)d_in[4];
  const float* w_ih        = (const float*)d_in[5];
  const float* w_hh        = (const float*)d_in[6];
  const float* b_ih        = (const float*)d_in[7];
  const float* b_hh        = (const float*)d_in[8];
  const float* w_tag       = (const float*)d_in[9];
  const float* b_tag       = (const float*)d_in[10];
  const float* start_trans = (const float*)d_in[11];
  const float* end_trans   = (const float*)d_in[12];
  const float* trans       = (const float*)d_in[13];

  char* ws = (char*)d_ws;
  __bf16* xbf   = (__bf16*)(ws + OFF_XBF);
  float*  g0    = (float*)(ws + OFF_G0);
  float*  g1    = (float*)(ws + OFF_G1);
  float*  feats = (float*)(ws + OFF_FEATS);
  float*  em    = (float*)(ws + OFF_EM);
  __bf16* wihb  = (__bf16*)(ws + OFF_WIH);
  __bf16* whhb  = (__bf16*)(ws + OFF_WHH);

  // weight + embedding prep
  k_cvt_w<<<3072, 256, 0, stream>>>(w_ih, w_hh, wihb, whhb);
  k_embed<<<(BB * SS * EMB) / 256, 256, 0, stream>>>(sentence, embed, xbf);

  for (int l = 0; l < 2; ++l) {
    if (l) k_f2b<<<(BB * SS * HID) / 256, 256, 0, stream>>>(feats, xbf, BB * SS * HID);
    for (int d = 0; d < 2; ++d) {
      k_gemm<<<2048, 256, 0, stream>>>(
          xbf, wihb + (size_t)(l * 2 + d) * GG * EMB,
          b_ih + (size_t)(l * 2 + d) * GG, b_hh + (size_t)(l * 2 + d) * GG,
          d ? g1 : g0);
    }
    k_lstm<<<2, 256, 0, stream>>>(g0, g1, whhb + (size_t)l * 2 * GG * H2, length, feats);
  }

  k_em<<<(BB * SS * NTAGS) / 256, 256, 0, stream>>>(feats, w_tag, b_tag, em);
  k_crf<<<1, BB, 0, stream>>>(em, tags, length, start_trans, end_trans, trans,
                              (float*)d_out);
}
